// MultiHeadAttention_39986145526232
// MI455X (gfx1250) — compile-verified, matrix operand order checked by reading, not run on hardware
//
#include <hip/hip_runtime.h>
#include <hip/hip_bf16.h>

#if __has_builtin(__builtin_amdgcn_global_load_async_to_lds_b128)
#define HAS_ASYNC 1
#else
#define HAS_ASYNC 0
#endif

#if __has_builtin(__builtin_amdgcn_tensor_load_to_lds) && \
    __has_builtin(__builtin_amdgcn_s_wait_tensorcnt)
#define HAS_TDM 1
#else
#define HAS_TDM 0
#endif

#define D_MODEL 1024
#define N_HEAD  16
#define BATCH   16
#define SEQ     1024
#define DFF     4096
#define ROWS    (BATCH*SEQ)   /* 16384 */

typedef __bf16 bf16_t;
typedef bf16_t v16bf __attribute__((ext_vector_type(16)));
typedef float  v8f   __attribute__((ext_vector_type(8)));

/* pointer/vector types matching the gfx1250 builtin prototypes */
typedef int          v4i __attribute__((vector_size(16)));
typedef unsigned int v4u __attribute__((vector_size(16)));
typedef int          v8i __attribute__((vector_size(32)));
typedef __attribute__((address_space(1))) v4i* as1_v4i_p;
typedef __attribute__((address_space(3))) v4i* as3_v4i_p;

union Frag { v16bf v; uint4 q[2]; unsigned short u[16]; };

__device__ __forceinline__ unsigned short f2bf(float f) {
  union { float f; unsigned int u; } x; x.f = f;
  unsigned int r = x.u + 0x7FFFu + ((x.u >> 16) & 1u);   // RNE
  return (unsigned short)(r >> 16);
}

__device__ __forceinline__ v8f v8f_zero() {
  v8f z;
#pragma unroll
  for (int i = 0; i < 8; ++i) z[i] = 0.0f;
  return z;
}

__device__ __forceinline__ v8f wmma_bf16(const Frag& a, const Frag& b, v8f c) {
  return __builtin_amdgcn_wmma_f32_16x16x32_bf16(false, a.v, false, b.v,
                                                 (short)0, c, false, false);
}

/* 16B global -> LDS copy: async DMA on gfx1250 if available, else via regs */
__device__ __forceinline__ void g2l_b128(const void* g, void* l) {
#if HAS_ASYNC
  __builtin_amdgcn_global_load_async_to_lds_b128(
      (as1_v4i_p)(v4i*)g, (as3_v4i_p)(v4i*)l, 0, 0);
#else
  *(uint4*)l = *(const uint4*)g;
#endif
}

__device__ __forceinline__ void wait_async() {
#if HAS_ASYNC
#if __has_builtin(__builtin_amdgcn_s_wait_asynccnt)
  __builtin_amdgcn_s_wait_asynccnt(0);
#else
  asm volatile("s_wait_asynccnt 0x0" ::: "memory");
#endif
#endif
}

/* LDS byte offset of a generic pointer into shared memory (AS3 ptrtoint) */
__device__ __forceinline__ unsigned lds_offset(const void* p) {
  return (unsigned)(size_t)(__attribute__((address_space(3))) const void*)p;
}

/* ---------------------------------------------------------------------
 * TDM: DMA one [128 rows x 32 bf16] tile of a row-major [rows,K] tensor
 * into LDS, padding each 64B row with 16B (-> 80B = LSTR*2 row stride).
 * D# group0/group1 packed per CDNA5 ISA 8.3/8.4; groups 2/3 zero (2D).
 * 6-arg builtin variant: (g0, g1, g2, g3, g_ext, cpol).
 * ------------------------------------------------------------------- */
__device__ __forceinline__ void tdm_load_tile(const void* gtile, unsigned lds_off,
                                              int K_elems, int rows) {
#if HAS_TDM
  unsigned long long ga = (unsigned long long)(size_t)gtile;
  v4u g0;
  g0[0] = 1u;                                   /* count=1 valid descriptor   */
  g0[1] = lds_off;                              /* lds_addr (bytes)           */
  g0[2] = (unsigned)(ga & 0xffffffffu);         /* global_addr[31:0]          */
  g0[3] = (unsigned)(ga >> 32) | (2u << 30);    /* global_addr[56:32], type=2 */
  v8i g1;
  g1[0] = (1 << 16)      /* data_size = 2 bytes            */
        | (1 << 20)      /* pad_enable                     */
        | (3 << 22)      /* pad_interval: 16 DWORDs (64B)  */
        | (3 << 25);     /* pad_amount:    4 DWORDs (16B)  */
  g1[1] = (K_elems & 0xffff) << 16;             /* tensor_dim0[15:0]          */
  g1[2] = ((K_elems >> 16) & 0xffff)            /* tensor_dim0[31:16]         */
        | ((rows & 0xffff) << 16);              /* tensor_dim1[15:0]          */
  g1[3] = ((rows >> 16) & 0xffff)               /* tensor_dim1[31:16]         */
        | (32 << 16);                           /* tile_dim0 = 32 elements    */
  g1[4] = 128;                                  /* tile_dim1 = 128 rows       */
  g1[5] = K_elems;                              /* tensor_dim0_stride[31:0]   */
  g1[6] = 0;                                    /* stride0[47:32], stride1 lo */
  g1[7] = 0;
  v4i z4 = {0, 0, 0, 0};
  v8i z8 = {0, 0, 0, 0, 0, 0, 0, 0};
  __builtin_amdgcn_tensor_load_to_lds(g0, g1, z4, z4, z8, 0);
#else
  (void)gtile; (void)lds_off; (void)K_elems; (void)rows;
#endif
}

__device__ __forceinline__ void wait_tensor() {
#if HAS_TDM
  __builtin_amdgcn_s_wait_tensorcnt(0);
#endif
}

/* ---------------- f32 -> bf16 convert (vectorized x4) ---------------- */
__global__ void __launch_bounds__(256)
cvt_bf16_kernel(const float* __restrict__ in, unsigned short* __restrict__ out,
                long long n4) {
  long long i = (long long)blockIdx.x * 256 + threadIdx.x;
  if (i >= n4) return;
  float4 v = ((const float4*)in)[i];
  union { unsigned short u[4]; uint2 q; } o;
  o.u[0] = f2bf(v.x); o.u[1] = f2bf(v.y); o.u[2] = f2bf(v.z); o.u[3] = f2bf(v.w);
  ((uint2*)out)[i] = o.q;
}

/* ---------- transpose + convert: out[C,R] bf16 = in[R,C]^T f32 ---------- */
__global__ void __launch_bounds__(256)
transpose_bf16_kernel(const float* __restrict__ in, unsigned short* __restrict__ out,
                      int R, int C) {
  __shared__ float tile[32][33];
  const int tx = threadIdx.x & 31;
  const int ty = threadIdx.x >> 5;            // 0..7
  const int c0 = blockIdx.x * 32;
  const int r0 = blockIdx.y * 32;
#pragma unroll
  for (int i = 0; i < 32; i += 8) {
    int r = r0 + ty + i, c = c0 + tx;
    tile[ty + i][tx] = in[(size_t)r * C + c];
  }
  __syncthreads();
#pragma unroll
  for (int i = 0; i < 32; i += 8) {
    int c = c0 + ty + i, r = r0 + tx;
    out[(size_t)c * R + r] = f2bf(tile[tx][ty + i]);
  }
}

/* ---------------------------------------------------------------------
 * Generic bf16 WMMA GEMM:  C[M,N] = A[M,K] * B[N,K]^T   (both K-contiguous)
 * Block tile 128x128, 8 waves -> each wave 64x32 (4x2 WMMA tiles).
 * LDS double-buffered. Tiles are filled by the Tensor Data Mover (one
 * D# descriptor per tile, issued by wave 0, TENSORcnt-tracked) with the
 * async global->LDS DMA path as fallback; the DMA fills tile kt+1 while
 * all 8 waves run WMMAs on tile kt.
 * EPI 0: outB = bf16(acc)
 * EPI 1: outF = acc + bias[col] + resid[row*N+col]
 * EPI 2: outB = bf16(relu(acc + bias[col]))
 * ------------------------------------------------------------------- */
#define BM 128
#define BN 128
#define TK 32
#define LSTR 40   /* padded LDS row stride in halves (64B data + 16B pad) */

template<int EPI>
__global__ void __launch_bounds__(256)
gemm_bf16_kernel(const unsigned short* __restrict__ A,
                 const unsigned short* __restrict__ B,
                 int M, int N, int K,
                 const float* __restrict__ bias,
                 const float* __restrict__ resid,
                 float* __restrict__ outF,
                 unsigned short* __restrict__ outB) {
  __shared__ __attribute__((aligned(16))) unsigned short As[2][BM][LSTR];
  __shared__ __attribute__((aligned(16))) unsigned short Bs[2][BN][LSTR];

  const int t    = threadIdx.x;
  const int lane = t & 31;
  const int w    = t >> 5;
  const int wm   = w >> 2;                 // 0..1
  const int wn   = w & 3;                  // 0..3
  const int n0   = blockIdx.x * BN;
  const int m0   = blockIdx.y * BM;

  const int srow  = t & 127;               // staging row (fallback path)
  const int shalf = t >> 7;                // 0/1 -> K offset 0/16 halves

  const unsigned short* arow = A + (size_t)(m0 + srow) * K + shalf * 16;
  const unsigned short* brow = B + (size_t)(n0 + srow) * K + shalf * 16;

  v8f acc[4][2];
#pragma unroll
  for (int i = 0; i < 4; ++i)
#pragma unroll
    for (int j = 0; j < 2; ++j) acc[i][j] = v8f_zero();

#if HAS_TDM
  auto stage = [&](int buf, int kt) {
    if (w == 0) {
      tdm_load_tile(A + (size_t)m0 * K + kt * TK, lds_offset(&As[buf][0][0]), K, M);
      tdm_load_tile(B + (size_t)n0 * K + kt * TK, lds_offset(&Bs[buf][0][0]), K, N);
    }
  };
#else
  auto stage = [&](int buf, int kt) {
    const unsigned short* ap = arow + kt * TK;
    const unsigned short* bp = brow + kt * TK;
    g2l_b128(ap,     &As[buf][srow][shalf * 16]);
    g2l_b128(ap + 8, &As[buf][srow][shalf * 16 + 8]);
    g2l_b128(bp,     &Bs[buf][srow][shalf * 16]);
    g2l_b128(bp + 8, &Bs[buf][srow][shalf * 16 + 8]);
  };
#endif

  stage(0, 0);
  const int KT = K / TK;
  for (int kt = 0; kt < KT; ++kt) {
#if HAS_TDM
    if (w == 0) wait_tensor();   /* wave 0 drains TDM, barrier publishes */
#else
    wait_async();                /* drain this wave's async fills        */
#endif
    __syncthreads();             /* all fills visible in LDS             */
    if (kt + 1 < KT) stage((kt + 1) & 1, kt + 1);
    if (kt + 2 < KT) {           /* gfx1250 global_prefetch_b8 */
      __builtin_prefetch(arow + (kt + 2) * TK, 0, 1);
      __builtin_prefetch(brow + (kt + 2) * TK, 0, 1);
    }

    const int buf  = kt & 1;
    const int koff = (lane >> 4) * 16;
    const int rr   = lane & 15;
    Frag af[4], bfr[2];
#pragma unroll
    for (int mt = 0; mt < 4; ++mt) {
      int r = wm * 64 + mt * 16 + rr;
      af[mt].q[0] = *(const uint4*)&As[buf][r][koff];
      af[mt].q[1] = *(const uint4*)&As[buf][r][koff + 8];
    }
#pragma unroll
    for (int nt = 0; nt < 2; ++nt) {
      int r = wn * 32 + nt * 16 + rr;
      bfr[nt].q[0] = *(const uint4*)&Bs[buf][r][koff];
      bfr[nt].q[1] = *(const uint4*)&Bs[buf][r][koff + 8];
    }
#pragma unroll
    for (int mt = 0; mt < 4; ++mt)
#pragma unroll
      for (int nt = 0; nt < 2; ++nt)
        acc[mt][nt] = wmma_bf16(af[mt], bfr[nt], acc[mt][nt]);
  }

  /* epilogue: C layout -> VGPR r holds (M = r + 8*(lane>>4), N = lane&15) */
  const int colb = lane & 15;
  const int rh   = (lane >> 4) * 8;
#pragma unroll
  for (int mt = 0; mt < 4; ++mt)
#pragma unroll
    for (int nt = 0; nt < 2; ++nt) {
      int col = n0 + wn * 32 + nt * 16 + colb;
#pragma unroll
      for (int r = 0; r < 8; ++r) {
        int row = m0 + wm * 64 + mt * 16 + rh + r;
        float v = acc[mt][nt][r];
        size_t idx = (size_t)row * N + col;
        if (EPI == 0) {
          outB[idx] = f2bf(v);
        } else if (EPI == 1) {
          outF[idx] = v + bias[col] + resid[idx];
        } else {
          v += bias[col];
          outB[idx] = f2bf(v > 0.0f ? v : 0.0f);
        }
      }
    }
  (void)M;
}

/* ---------------------------------------------------------------------
 * Attention per (b,h):  G = q^T k (WMMA, K=S), s2 = U_h G / 8, softmax,
 * ctx = attn v^T (WMMA, N=S), scattered into ctxm[(u*16+h), s].
 * ------------------------------------------------------------------- */
__global__ void __launch_bounds__(256)
attention_kernel(const unsigned short* __restrict__ qm,
                 const unsigned short* __restrict__ km,
                 const unsigned short* __restrict__ vm,
                 const float* __restrict__ U,
                 unsigned short* __restrict__ ctxm) {
  __shared__ __attribute__((aligned(16))) unsigned short qT[64][LSTR];
  __shared__ __attribute__((aligned(16))) unsigned short kT[64][LSTR];
  __shared__ __attribute__((aligned(16))) float Gs[64][68];
  __shared__ __attribute__((aligned(16))) float us[64][64];
  __shared__ __attribute__((aligned(16))) float s2s[64][64];
  __shared__ __attribute__((aligned(16))) unsigned short attnS[64][64];

  const int t    = threadIdx.x;
  const int lane = t & 31;
  const int w    = t >> 5;
  const int bi   = blockIdx.x >> 4;
  const int h    = blockIdx.x & 15;
  const size_t baseQ = (size_t)bi * SEQ * D_MODEL + h * 64;

  /* stage U_h into LDS (f32) via async DMA; drained at end of phase 1 */
  {
    int l = t >> 2, cb = (t & 3) * 16;
    const float* up = U + (size_t)(bi * 64 + l) * D_MODEL + h * 64 + cb;
#pragma unroll
    for (int i = 0; i < 4; ++i) g2l_b128(up + i * 4, &us[l][cb + i * 4]);
  }

  /* phase 1: G = q_h^T k_h, 16 WMMA tiles, 2 per wave */
  v8f accG0 = v8f_zero(), accG1 = v8f_zero();
  const int ti0 = 2 * w;
  for (int c = 0; c < SEQ / 32; ++c) {
    __syncthreads();
    {   /* stage transposed 32x64 chunks of q,k */
      int sr = t >> 3, cb = (t & 7) * 8;
      union { uint4 q; unsigned short u[8]; } vq, vk;
      const unsigned short* qp = qm + baseQ + (size_t)(c * 32 + sr) * D_MODEL + cb;
      const unsigned short* kp = km + baseQ + (size_t)(c * 32 + sr) * D_MODEL + cb;
      vq.q = *(const uint4*)qp;
      vk.q = *(const uint4*)kp;
#pragma unroll
      for (int i = 0; i < 8; ++i) { qT[cb + i][sr] = vq.u[i]; kT[cb + i][sr] = vk.u[i]; }
    }
    __syncthreads();
    const int koff = (lane >> 4) * 16;
    const int rr   = lane & 15;
#pragma unroll
    for (int s = 0; s < 2; ++s) {
      int ti = ti0 + s, mi = ti >> 2, ni = ti & 3;
      Frag a, b;
      a.q[0] = *(const uint4*)&qT[mi * 16 + rr][koff];
      a.q[1] = *(const uint4*)&qT[mi * 16 + rr][koff + 8];
      b.q[0] = *(const uint4*)&kT[ni * 16 + rr][koff];
      b.q[1] = *(const uint4*)&kT[ni * 16 + rr][koff + 8];
      if (s == 0) accG0 = wmma_bf16(a, b, accG0);
      else        accG1 = wmma_bf16(a, b, accG1);
    }
  }
  {
    const int rr = lane & 15, rh = (lane >> 4) * 8;
    int mi = ti0 >> 2, ni = ti0 & 3;
#pragma unroll
    for (int r = 0; r < 8; ++r) Gs[mi * 16 + rh + r][ni * 16 + rr] = accG0[r];
    mi = (ti0 + 1) >> 2; ni = (ti0 + 1) & 3;
#pragma unroll
    for (int r = 0; r < 8; ++r) Gs[mi * 16 + rh + r][ni * 16 + rr] = accG1[r];
  }
  wait_async();          /* us[] fills complete before phase 2 reads */
  __syncthreads();

  /* phase 2: s2 = U_h @ G * (1/sqrt(64)) */
  {
    int l = t >> 2, db = (t & 3) * 16;
    for (int d = db; d < db + 16; ++d) {
      float a = 0.0f;
#pragma unroll 8
      for (int j = 0; j < 64; ++j) a += us[l][j] * Gs[j][d];
      s2s[l][d] = a * 0.125f;
    }
  }
  __syncthreads();

  /* softmax over d, one thread per row */
  if (t < 64) {
    float mx = -1e30f;
    for (int d = 0; d < 64; ++d) mx = fmaxf(mx, s2s[t][d]);
    float sum = 0.0f;
    for (int d = 0; d < 64; ++d) { float e = __expf(s2s[t][d] - mx); s2s[t][d] = e; sum += e; }
    float inv = 1.0f / sum;
    for (int d = 0; d < 64; ++d) attnS[t][d] = f2bf(s2s[t][d] * inv);
  }
  __syncthreads();

  /* phase 3: ctx = attn @ v^T, M=64, K=64, N=1024 */
  {
    const int rr = lane & 15;
    const int koff = (lane >> 4) * 16;
    const int rh = (lane >> 4) * 8;
    Frag aF[4][2];
#pragma unroll
    for (int mt = 0; mt < 4; ++mt)
#pragma unroll
      for (int ks = 0; ks < 2; ++ks) {
        aF[mt][ks].q[0] = *(const uint4*)&attnS[mt * 16 + rr][ks * 32 + koff];
        aF[mt][ks].q[1] = *(const uint4*)&attnS[mt * 16 + rr][ks * 32 + koff + 8];
      }
    for (int nt = w; nt < 64; nt += 8) {
      Frag b0, b1;
      const unsigned short* vp = vm + baseQ + (size_t)(nt * 16 + rr) * D_MODEL;
      b0.q[0] = *(const uint4*)(vp + koff);
      b0.q[1] = *(const uint4*)(vp + koff + 8);
      b1.q[0] = *(const uint4*)(vp + 32 + koff);
      b1.q[1] = *(const uint4*)(vp + 32 + koff + 8);
      v8f cacc[4];
#pragma unroll
      for (int mt = 0; mt < 4; ++mt) cacc[mt] = v8f_zero();
#pragma unroll
      for (int mt = 0; mt < 4; ++mt) {
        cacc[mt] = wmma_bf16(aF[mt][0], b0, cacc[mt]);
        cacc[mt] = wmma_bf16(aF[mt][1], b1, cacc[mt]);
      }
#pragma unroll
      for (int mt = 0; mt < 4; ++mt)
#pragma unroll
        for (int r = 0; r < 8; ++r) {
          int l = mt * 16 + rh + r;
          ctxm[(size_t)(bi * 1024 + l * 16 + h) * 1024 + nt * 16 + rr] = f2bf(cacc[mt][r]);
        }
    }
  }
}

/* ------------------- LayerNorm (block per row of 1024) ------------------- */
template<bool WBF>
__global__ void __launch_bounds__(256)
layernorm_kernel(const float* __restrict__ X, const float* __restrict__ g,
                 const float* __restrict__ b, float* __restrict__ outF,
                 unsigned short* __restrict__ outB) {
  __shared__ float r1[256], r2[256];
  const int row = blockIdx.x, t = threadIdx.x;
  const float* x = X + (size_t)row * D_MODEL;
  float v[4], s = 0.0f, ss = 0.0f;
#pragma unroll
  for (int i = 0; i < 4; ++i) { v[i] = x[t + 256 * i]; s += v[i]; ss += v[i] * v[i]; }
  r1[t] = s; r2[t] = ss;
  __syncthreads();
  for (int off = 128; off > 0; off >>= 1) {
    if (t < off) { r1[t] += r1[t + off]; r2[t] += r2[t + off]; }
    __syncthreads();
  }
  const float mean = r1[0] * (1.0f / D_MODEL);
  const float var  = r2[0] * (1.0f / D_MODEL) - mean * mean;
  const float inv  = rsqrtf(var + 1e-6f);
#pragma unroll
  for (int i = 0; i < 4; ++i) {
    int c = t + 256 * i;
    float y = (v[i] - mean) * inv * g[c] + b[c];
    size_t idx = (size_t)row * D_MODEL + c;
    outF[idx] = y;
    if (WBF) outB[idx] = f2bf(y);
  }
}

/* ------------------------------- driver ------------------------------- */
extern "C" void kernel_launch(void* const* d_in, const int* in_sizes, int n_in,
                              void* d_out, int out_size, void* d_ws, size_t ws_size,
                              hipStream_t stream) {
  (void)in_sizes; (void)n_in; (void)out_size; (void)ws_size;
  const float* Q    = (const float*)d_in[0];
  const float* Kin  = (const float*)d_in[1];
  const float* Vin  = (const float*)d_in[2];
  const float* U    = (const float*)d_in[3];
  const float* W_Q  = (const float*)d_in[4];
  const float* W_K  = (const float*)d_in[5];
  const float* W_V  = (const float*)d_in[6];
  const float* W_O  = (const float*)d_in[7];
  const float* b_O  = (const float*)d_in[8];
  const float* ln_g = (const float*)d_in[9];
  const float* ln_b = (const float*)d_in[10];
  const float* W1   = (const float*)d_in[11];
  const float* b1   = (const float*)d_in[12];
  const float* W2   = (const float*)d_in[13];
  const float* b2   = (const float*)d_in[14];
  const float* g2   = (const float*)d_in[15];
  const float* bt2  = (const float*)d_in[16];
  float* out = (float*)d_out;

  char* p = (char*)d_ws;
  auto alloc = [&](size_t bytes) -> char* {
    char* r = p; p += (bytes + 255) & ~(size_t)255; return r;
  };
  const size_t MD = (size_t)ROWS * D_MODEL;
  const size_t DD = (size_t)D_MODEL * D_MODEL;
  unsigned short* Qb   = (unsigned short*)alloc(MD * 2);
  unsigned short* Kb   = (unsigned short*)alloc(MD * 2);
  unsigned short* Vb   = (unsigned short*)alloc(MD * 2);
  unsigned short* Wqb  = (unsigned short*)alloc(DD * 2);
  unsigned short* Wkb  = (unsigned short*)alloc(DD * 2);
  unsigned short* Wvb  = (unsigned short*)alloc(DD * 2);
  unsigned short* Wob  = (unsigned short*)alloc(DD * 2);
  unsigned short* W1t  = (unsigned short*)alloc((size_t)DFF * D_MODEL * 2);
  unsigned short* W2t  = (unsigned short*)alloc((size_t)D_MODEL * DFF * 2);
  unsigned short* qm   = (unsigned short*)alloc(MD * 2);
  unsigned short* km   = (unsigned short*)alloc(MD * 2);
  unsigned short* vmm  = (unsigned short*)alloc(MD * 2);
  unsigned short* ctxm = (unsigned short*)alloc(MD * 2);
  float*          X1   = (float*)alloc(MD * 4);
  float*          Xln  = (float*)alloc(MD * 4);
  unsigned short* Xlnb = (unsigned short*)alloc(MD * 2);
  unsigned short* Hb   = (unsigned short*)alloc((size_t)ROWS * DFF * 2);
  float*          Y    = (float*)alloc(MD * 4);

  auto cvt = [&](const float* src, unsigned short* dst, size_t n) {
    long long n4 = (long long)(n / 4);
    cvt_bf16_kernel<<<dim3((unsigned)((n4 + 255) / 256)), dim3(256), 0, stream>>>(src, dst, n4);
  };
  cvt(Q, Qb, MD); cvt(Kin, Kb, MD); cvt(Vin, Vb, MD);
  cvt(W_Q, Wqb, DD); cvt(W_K, Wkb, DD); cvt(W_V, Wvb, DD); cvt(W_O, Wob, DD);

  transpose_bf16_kernel<<<dim3(DFF / 32, D_MODEL / 32), dim3(256), 0, stream>>>(W1, W1t, D_MODEL, DFF);
  transpose_bf16_kernel<<<dim3(D_MODEL / 32, DFF / 32), dim3(256), 0, stream>>>(W2, W2t, DFF, D_MODEL);

  dim3 gProj(D_MODEL / BN, ROWS / BM);
  gemm_bf16_kernel<0><<<gProj, dim3(256), 0, stream>>>(Qb, Wqb, ROWS, D_MODEL, D_MODEL,
                                                       nullptr, nullptr, nullptr, qm);
  gemm_bf16_kernel<0><<<gProj, dim3(256), 0, stream>>>(Kb, Wkb, ROWS, D_MODEL, D_MODEL,
                                                       nullptr, nullptr, nullptr, km);
  gemm_bf16_kernel<0><<<gProj, dim3(256), 0, stream>>>(Vb, Wvb, ROWS, D_MODEL, D_MODEL,
                                                       nullptr, nullptr, nullptr, vmm);

  attention_kernel<<<dim3(BATCH * N_HEAD), dim3(256), 0, stream>>>(qm, km, vmm, U, ctxm);

  gemm_bf16_kernel<1><<<gProj, dim3(256), 0, stream>>>(ctxm, Wob, ROWS, D_MODEL, D_MODEL,
                                                       b_O, Q, X1, nullptr);
  layernorm_kernel<true><<<dim3(ROWS), dim3(256), 0, stream>>>(X1, ln_g, ln_b, Xln, Xlnb);

  dim3 gF1(DFF / BN, ROWS / BM);
  gemm_bf16_kernel<2><<<gF1, dim3(256), 0, stream>>>(Xlnb, W1t, ROWS, DFF, D_MODEL,
                                                     b1, nullptr, nullptr, Hb);
  gemm_bf16_kernel<1><<<gProj, dim3(256), 0, stream>>>(Hb, W2t, ROWS, D_MODEL, DFF,
                                                       b2, Xln, Y, nullptr);
  layernorm_kernel<false><<<dim3(ROWS), dim3(256), 0, stream>>>(Y, g2, bt2, out, nullptr);
}
